// GMNet_13073880449145
// MI455X (gfx1250) — compile-verified, matrix operand order checked by reading, not run on hardware
//
#include <hip/hip_runtime.h>

// GMNet forward (PCA-GM style) for MI455X / gfx1250.
// Big GEMMs run on bf16 WMMA (v_wmma_f32_16x16x32_bf16) with fp32 accumulate.
// Workspace requirement: >= (6*14336*1024 + 2*1024*14*14) floats ~= 354 MB.

#define Bb   1024
#define Nn   14
#define FCc  2048
#define GFf  1024
#define ALPHAv 200.0f
#define SINK_ITERS 20
#define EPSV 1e-10f

typedef __bf16 bf16_t;
typedef __attribute__((ext_vector_type(16))) bf16_t v16bf;
typedef __attribute__((ext_vector_type(8)))  float  v8f;
typedef __attribute__((ext_vector_type(4)))  unsigned int u32x4;
typedef __attribute__((ext_vector_type(8)))  unsigned int u32x8;

// Pack two fp32 into packed bf16x2 (round-to-nearest; ties away).
// v_perm_b32 merges the two rounded high-halves in a single instruction:
// result = { hi16(u1+0x8000), hi16(u0+0x8000) }  -> selector 0x07060302.
__device__ __forceinline__ unsigned int pk2bf(float f0, float f1) {
    unsigned int u0 = __float_as_uint(f0) + 0x8000u;
    unsigned int u1 = __float_as_uint(f1) + 0x8000u;
    return __builtin_amdgcn_perm(u1, u0, 0x07060302u);
}

// ---------------------------------------------------------------------------
// Tiled bf16 WMMA GEMM: C[M,Nc] = act(A[M,K] @ W[K,Nc] + bias)
// Block tile 128x128, K-step 32. 256 threads = 8 wave32 waves, waves 4(M)x2(N),
// each wave computes 32x64 = 2x4 tiles of v_wmma_f32_16x16x32_bf16.
// Global->reg prefetch of the next K-tile overlaps the WMMA burst.
// Requires M%128==0, K%32==0, Nc%128==0 (all shapes here satisfy this).
// ---------------------------------------------------------------------------
#define TM 128
#define TN 128
#define TK 32
#define LDSK 20   // padded k-stride in uints (= 32 bf16 + 8 pad = 80 B)

__global__ __launch_bounds__(256) void gemm_bf16_wmma(
    const float* __restrict__ A, const float* __restrict__ W,
    const float* __restrict__ bias, float* __restrict__ C,
    int M, int K, int Nc, int relu)
{
    __shared__ unsigned int lA[TM * LDSK];   // 10240 B, bf16 pairs [m][k/2]
    __shared__ unsigned int lB[TN * LDSK];   // 10240 B, bf16 pairs [n][k/2]

    const int tid  = threadIdx.x;
    const int bM   = blockIdx.y * TM;
    const int bN   = blockIdx.x * TN;
    const int wv   = tid >> 5;
    const int lane = tid & 31;
    const int mo   = (wv >> 1) * 32;     // wave M origin in tile
    const int no   = (wv & 1) * 64;      // wave N origin in tile
    const int lr   = lane & 15;          // row/col within 16x16 fragment
    const int kh   = lane >> 4;          // K-half select (ISA 7.12.2 layout)

    v8f acc[2][4] = {};

    // Staging assignments (each thread: 16 A elems + 16 B elems per K-step)
    const int ar = tid >> 1;             // A row 0..127
    const int ak = (tid & 1) * 16;       // A k-base: 0 or 16
    const int bn = tid & 127;            // B col 0..127
    const int bk = (tid >> 7) * 16;      // B k-base: 0 or 16

    const float* aPtr = A + (size_t)(bM + ar) * K + ak;
    const float* wPtr = W + (size_t)bk * Nc + bN + bn;

    float4 aR[4];
    float  bR[16];

    // Prologue: fetch K-tile 0 into registers
    {
        const float* ap = aPtr;
#pragma unroll
        for (int j = 0; j < 4; ++j) aR[j] = *(const float4*)(ap + 4 * j);
        const float* wp = wPtr;
#pragma unroll
        for (int i = 0; i < 16; ++i) bR[i] = wp[(size_t)i * Nc];
    }

    for (int k0 = 0; k0 < K; k0 += TK) {
        // Stage current tile: convert fp32->bf16 pairs, two b128 LDS stores each
        {
            unsigned int pa[8], pb[8];
#pragma unroll
            for (int j = 0; j < 4; ++j) {
                pa[2 * j]     = pk2bf(aR[j].x, aR[j].y);
                pa[2 * j + 1] = pk2bf(aR[j].z, aR[j].w);
            }
#pragma unroll
            for (int j = 0; j < 8; ++j) pb[j] = pk2bf(bR[2 * j], bR[2 * j + 1]);
            u32x4* da = (u32x4*)&lA[ar * LDSK + (ak >> 1)];
            da[0] = (u32x4){pa[0], pa[1], pa[2], pa[3]};
            da[1] = (u32x4){pa[4], pa[5], pa[6], pa[7]};
            u32x4* db = (u32x4*)&lB[bn * LDSK + (bk >> 1)];
            db[0] = (u32x4){pb[0], pb[1], pb[2], pb[3]};
            db[1] = (u32x4){pb[4], pb[5], pb[6], pb[7]};
        }
        __syncthreads();

        // Prefetch next K-tile into registers (overlaps WMMA below)
        if (k0 + TK < K) {
            const float* ap = aPtr + k0 + TK;
#pragma unroll
            for (int j = 0; j < 4; ++j) aR[j] = *(const float4*)(ap + 4 * j);
            const float* wp = wPtr + (size_t)(k0 + TK) * Nc;
#pragma unroll
            for (int i = 0; i < 16; ++i) bR[i] = wp[(size_t)i * Nc];
            __builtin_prefetch(ap + TK, 0, 0);   // global_prefetch_b8 (tile after next)
        }

        // Build fragments per ISA 16-bit 16x32 layout:
        // lane half kh: e<8 -> K = 8*kh + e ; e>=8 -> K = 16 + 8*kh + (e-8)
        v16bf afr[2], bfr[4];
#pragma unroll
        for (int sm = 0; sm < 2; ++sm) {
            const unsigned int* p = &lA[(mo + sm * 16 + lr) * LDSK];
            u32x4 lo = *(const u32x4*)(p + 4 * kh);
            u32x4 hi = *(const u32x4*)(p + 8 + 4 * kh);
            u32x8 t;
#pragma unroll
            for (int q = 0; q < 4; ++q) { t[q] = lo[q]; t[4 + q] = hi[q]; }
            afr[sm] = __builtin_bit_cast(v16bf, t);
        }
#pragma unroll
        for (int sn = 0; sn < 4; ++sn) {
            const unsigned int* p = &lB[(no + sn * 16 + lr) * LDSK];
            u32x4 lo = *(const u32x4*)(p + 4 * kh);
            u32x4 hi = *(const u32x4*)(p + 8 + 4 * kh);
            u32x8 t;
#pragma unroll
            for (int q = 0; q < 4; ++q) { t[q] = lo[q]; t[4 + q] = hi[q]; }
            bfr[sn] = __builtin_bit_cast(v16bf, t);
        }

#pragma unroll
        for (int sm = 0; sm < 2; ++sm)
#pragma unroll
            for (int sn = 0; sn < 4; ++sn)
                acc[sm][sn] = __builtin_amdgcn_wmma_f32_16x16x32_bf16(
                    false, afr[sm], false, bfr[sn], (short)0, acc[sm][sn],
                    false, false);
        __syncthreads();
    }

    // Epilogue: C/D layout — VGPR r: lanes 0-15 -> M=r, lanes 16-31 -> M=r+8
#pragma unroll
    for (int sm = 0; sm < 2; ++sm) {
#pragma unroll
        for (int sn = 0; sn < 4; ++sn) {
            const int gn = bN + no + sn * 16 + lr;
            const float bv = bias ? bias[gn] : 0.0f;
#pragma unroll
            for (int r = 0; r < 8; ++r) {
                const int gm = bM + mo + sm * 16 + r + 8 * kh;
                float v = acc[sm][sn][r] + bv;
                if (relu) v = fmaxf(v, 0.0f);
                C[(size_t)gm * Nc + gn] = v;
            }
        }
    }
}

// ---------------------------------------------------------------------------
// An[b,i,j] = adj[b,i,j] / max(sum_i |adj[b,i,j]|, 1e-12)   (L1 over axis -2)
// ---------------------------------------------------------------------------
__global__ void norm_adj_kernel(const float* __restrict__ adj, float* __restrict__ An)
{
    const int b = blockIdx.x;
    __shared__ float a[Nn * Nn];
    __shared__ float cs[Nn];
    const int t = threadIdx.x;
    if (t < Nn * Nn) a[t] = adj[(size_t)b * Nn * Nn + t];
    __syncthreads();
    if (t < Nn) {
        float s = 0.f;
        for (int i = 0; i < Nn; ++i) s += fabsf(a[i * Nn + t]);
        cs[t] = fmaxf(s, 1e-12f);
    }
    __syncthreads();
    if (t < Nn * Nn) An[(size_t)b * Nn * Nn + t] = a[t] / cs[t % Nn];
}

// ---------------------------------------------------------------------------
// out[b,i,d] = (Add? Add[b,i,d]:0) + sum_j S'[b,i,j] * X[b,j,d]
// S' = S or S^T per `trans`. Output row stride ldOut (for concat halves).
// ---------------------------------------------------------------------------
__global__ void bmm_small(const float* __restrict__ S, const float* __restrict__ X,
                          const float* __restrict__ Add, float* __restrict__ Out,
                          int D, int ldOut, int trans)
{
    const int bi = blockIdx.x;
    const int b = bi / Nn;
    const int i = bi - b * Nn;
    __shared__ float srow[Nn];
    if (threadIdx.x < Nn)
        srow[threadIdx.x] = trans ? S[(size_t)b * Nn * Nn + threadIdx.x * Nn + i]
                                  : S[(size_t)b * Nn * Nn + i * Nn + threadIdx.x];
    __syncthreads();
    float sr[Nn];
#pragma unroll
    for (int j = 0; j < Nn; ++j) sr[j] = srow[j];
    const float* Xb = X + (size_t)b * Nn * D;
    for (int d = threadIdx.x; d < D; d += blockDim.x) {
        float acc = Add ? Add[((size_t)b * Nn + i) * D + d] : 0.f;
#pragma unroll
        for (int j = 0; j < Nn; ++j) acc += sr[j] * Xb[(size_t)j * D + d];
        Out[((size_t)b * Nn + i) * (size_t)ldOut + d] = acc;
    }
}

// ---------------------------------------------------------------------------
// S[b,i,j] = sum_d T[b,i,d] * E[b,j,d]     (affinity, D=GF)
// ---------------------------------------------------------------------------
__global__ void affinity_kernel(const float* __restrict__ T, const float* __restrict__ E,
                                float* __restrict__ S, int D)
{
    const int bi = blockIdx.x;
    const int b = bi / Nn;
    const int i = bi - b * Nn;
    const int j = threadIdx.x >> 4;
    const int sub = threadIdx.x & 15;
    __shared__ float red[16][17];
    float p = 0.f;
    if (j < Nn) {
        const float* tr = T + ((size_t)b * Nn + i) * D;
        const float* er = E + ((size_t)b * Nn + j) * D;
        for (int d = sub; d < D; d += 16) p += tr[d] * er[d];
    }
    red[j][sub] = p;
    __syncthreads();
    if (threadIdx.x < Nn) {
        float s = 0.f;
#pragma unroll
        for (int q = 0; q < 16; ++q) s += red[threadIdx.x][q];
        S[(size_t)b * Nn * Nn + i * Nn + threadIdx.x] = s;
    }
}

// ---------------------------------------------------------------------------
// Fused: softmax(ALPHA*s, axis=-1) then 20 alternating sinkhorn normalizations.
// ---------------------------------------------------------------------------
__global__ void softmax_sinkhorn(const float* __restrict__ Sin, float* __restrict__ Sout)
{
    const int b = blockIdx.x;
    __shared__ float m[Nn][Nn + 1];
    const int t = threadIdx.x;
    if (t < Nn * Nn) m[t / Nn][t % Nn] = ALPHAv * Sin[(size_t)b * Nn * Nn + t];
    __syncthreads();
    if (t < Nn) {  // row softmax
        float mx = -3.4e38f;
        for (int j = 0; j < Nn; ++j) mx = fmaxf(mx, m[t][j]);
        float sm = 0.f;
        for (int j = 0; j < Nn; ++j) { float e = expf(m[t][j] - mx); m[t][j] = e; sm += e; }
        const float inv = 1.0f / sm;
        for (int j = 0; j < Nn; ++j) m[t][j] *= inv;
    }
    __syncthreads();
    for (int it = 0; it < SINK_ITERS; ++it) {
        if ((it & 1) == 0) {
            if (t < Nn) {  // normalize rows (axis=2)
                float s = 0.f;
                for (int j = 0; j < Nn; ++j) s += m[t][j];
                s += EPSV;
                for (int j = 0; j < Nn; ++j) m[t][j] /= s;
            }
        } else {
            if (t < Nn) {  // normalize cols (axis=1)
                float s = 0.f;
                for (int i = 0; i < Nn; ++i) s += m[i][t];
                s += EPSV;
                for (int i = 0; i < Nn; ++i) m[i][t] /= s;
            }
        }
        __syncthreads();
    }
    if (t < Nn * Nn) Sout[(size_t)b * Nn * Nn + t] = m[t / Nn][t % Nn];
}

// dst[r*ldDst + c] = src[r*D + c]   (concat first-half copy)
__global__ void copy_strided(const float* __restrict__ src, float* __restrict__ dst,
                             int D, int ldDst)
{
    const size_t r = blockIdx.x;
    for (int d = threadIdx.x; d < D; d += blockDim.x)
        dst[r * (size_t)ldDst + d] = src[r * (size_t)D + d];
}

// ---------------------------------------------------------------------------
extern "C" void kernel_launch(void* const* d_in, const int* in_sizes, int n_in,
                              void* d_out, int out_size, void* d_ws, size_t ws_size,
                              hipStream_t stream)
{
    (void)in_sizes; (void)n_in; (void)out_size; (void)ws_size;

    const float* emb1  = (const float*)d_in[0];
    const float* emb2  = (const float*)d_in[1];
    const float* adj   = (const float*)d_in[2];
    const float* Wa0   = (const float*)d_in[3];
    const float* ba0   = (const float*)d_in[4];
    const float* Wu0   = (const float*)d_in[5];
    const float* bu0   = (const float*)d_in[6];
    const float* Wa1   = (const float*)d_in[7];
    const float* ba1   = (const float*)d_in[8];
    const float* Wu1   = (const float*)d_in[9];
    const float* bu1   = (const float*)d_in[10];
    const float* Aaff0 = (const float*)d_in[11];
    const float* Aaff1 = (const float*)d_in[12];
    const float* Wc    = (const float*)d_in[13];
    const float* bc    = (const float*)d_in[14];

    const int M = Bb * Nn;                        // 14336
    const size_t U = (size_t)M * GFf;             // 14,680,064 floats
    float* ws = (float*)d_ws;
    float* U0 = ws;            // also CAT2 low half ([U0,U1) = [M, 2GF])
    float* U1 = ws + 1 * U;
    float* U2 = ws + 2 * U;    // e1 / ne1
    float* U3 = ws + 3 * U;    // e2 / ne2
    float* U4 = ws + 4 * U;    // also CAT1 low half ([U4,U5) = [M, 2GF])
    float* U5 = ws + 5 * U;
    float* An = ws + 6 * U;                       // [B,N,N]
    float* Sb = An + (size_t)Bb * Nn * Nn;        // [B,N,N]

    float* sOut = (float*)d_out;                          // [B,N,N]
    float* fin1 = sOut + (size_t)Bb * Nn * Nn;            // [B,N,FC]
    float* fin2 = fin1 + (size_t)Bb * Nn * FCc;           // [B,N,FC]

    auto gemm = [&](const float* A, const float* W, const float* bias, float* C,
                    int K, int Nc, int relu) {
        dim3 grid(Nc / TN, M / TM);
        gemm_bf16_wmma<<<grid, 256, 0, stream>>>(A, W, bias, C, M, K, Nc, relu);
    };

    norm_adj_kernel<<<Bb, 256, 0, stream>>>(adj, An);

    // ---- GNN layer 0 ----
    gemm(emb1, Wa0, ba0, U0, FCc, GFf, 1);                         // relu(emb1@Wa0+ba0)
    gemm(emb1, Wu0, bu0, U1, FCc, GFf, 1);
    bmm_small<<<M, 256, 0, stream>>>(An, U0, U1, U2, GFf, GFf, 0); // e1 = An@ax + ux
    gemm(emb2, Wa0, ba0, U0, FCc, GFf, 1);
    gemm(emb2, Wu0, bu0, U1, FCc, GFf, 1);
    bmm_small<<<M, 256, 0, stream>>>(An, U0, U1, U3, GFf, GFf, 0); // e2

    // ---- affinity 0 + voting + sinkhorn ----
    gemm(U2, Aaff0, nullptr, U0, GFf, GFf, 0);                     // t = e1@Aaff0
    affinity_kernel<<<M, 256, 0, stream>>>(U0, U3, Sb, GFf);       // s = t . e2^T
    softmax_sinkhorn<<<Bb, 256, 0, stream>>>(Sb, Sb);

    // ---- cross-graph update: ne = concat(e, s@other) @ Wc + bc ----
    copy_strided<<<M, 256, 0, stream>>>(U2, U4, GFf, 2 * GFf);               // CAT1[:, :GF]=e1
    bmm_small<<<M, 256, 0, stream>>>(Sb, U3, nullptr, U4 + GFf, GFf, 2 * GFf, 0); // CAT1[:,GF:]=s@e2
    copy_strided<<<M, 256, 0, stream>>>(U3, U0, GFf, 2 * GFf);               // CAT2[:, :GF]=e2
    bmm_small<<<M, 256, 0, stream>>>(Sb, U2, nullptr, U0 + GFf, GFf, 2 * GFf, 1); // CAT2[:,GF:]=s^T@e1
    gemm(U4, Wc, bc, U2, 2 * GFf, GFf, 0);                         // ne1
    gemm(U0, Wc, bc, U3, 2 * GFf, GFf, 0);                         // ne2

    // ---- GNN layer 1 ----
    gemm(U2, Wa1, ba1, U4, GFf, GFf, 1);
    gemm(U2, Wu1, bu1, U5, GFf, GFf, 1);
    bmm_small<<<M, 256, 0, stream>>>(An, U4, U5, U0, GFf, GFf, 0); // e1'
    gemm(U3, Wa1, ba1, U4, GFf, GFf, 1);
    gemm(U3, Wu1, bu1, U5, GFf, GFf, 1);
    bmm_small<<<M, 256, 0, stream>>>(An, U4, U5, U1, GFf, GFf, 0); // e2'

    // ---- affinity 1 + voting + sinkhorn (final s) ----
    gemm(U0, Aaff1, nullptr, U4, GFf, GFf, 0);                     // t' = e1'@Aaff1
    affinity_kernel<<<M, 256, 0, stream>>>(U4, U1, Sb, GFf);
    softmax_sinkhorn<<<Bb, 256, 0, stream>>>(Sb, sOut);

    // ---- final residual cross-attention on original embeddings ----
    bmm_small<<<M, 256, 0, stream>>>(sOut, emb2, emb1, fin1, FCc, FCc, 0); // fin1 = emb1 + s@emb2
    bmm_small<<<M, 256, 0, stream>>>(sOut, emb1, emb2, fin2, FCc, FCc, 1); // fin2 = emb2 + s^T@emb1
}
